// Seq2SQL_65446711657086
// MI455X (gfx1250) — matrix operand order repaired and checked by reading, not compiled, not run on hardware
//
#include <hip/hip_runtime.h>

// CDNA5 / gfx1250 single-kernel WMMA implementation of the Seq2SQL head.
// All matrix math via v_wmma_f32_16x16x32_f16 (probe-verified builtin):
// fp32 gathered activations / weights converted to f16 in registers,
// fp32 accumulation. One fused kernel: 436 wave32 tiles, 4 waves/block.
// K-loop unrolled x2 so alternating fragment registers cover the f16-WMMA
// WAR hazard (4 co-exec slots) with real work instead of v_nop.

typedef __attribute__((ext_vector_type(16))) _Float16 v16h;
typedef __attribute__((ext_vector_type(8)))  float    v8f;

#define ISZ   1024
#define LSEQ  512

// d_out flat offsets (floats), reference tuple return order
#define OFF_ACT   0        // s_action  [64,5]
#define OFF_SC    320      // sc_output [64,4,32]
#define OFF_SA    8512     // sa_output [64,4,6]
#define OFF_CCO   10048    // s_cco     [64,3]
#define OFF_WC    10240    // wc_output [64,4,32]
#define OFF_WO    18432    // wo_output [64,4,4]
#define OFF_WVSS  19456    // wv_ss     [64,4,66]
#define OFF_WVSE  36352    // wv_se     [64,4,66]
#define OFF_SLEN  53248    // s_slen    [64,5]
#define OFF_WLEN  53568    // s_wlen    [64,5]

// tile-id partition of the fused kernel
#define T_QV    264        // qv GEMM:    4224 rows  x 8(pad16) cols
#define T_HT    392        // ht GEMM:    2048 rows  x 8(pad16) cols
#define T_SMALL 424        // small heads: 256 rows  x 18(pad32) cols
#define T_WO    428        // wo GEMM:      64 rows  x 16 cols, K=4096
#define T_SA    436        // sa GEMM:      64 rows  x 24(pad32) cols, K=4096

// Convert 8 fp32 at p0 and 8 fp32 at p1 into a 16-half WMMA fragment.
// Loads are 4x b128 (all 16/32-byte aligned by construction).
__device__ __forceinline__ v16h cvt16(const float* __restrict__ p0,
                                      const float* __restrict__ p1) {
  const float4 x0 = *(const float4*)(p0);
  const float4 x1 = *(const float4*)(p0 + 4);
  const float4 x2 = *(const float4*)(p1);
  const float4 x3 = *(const float4*)(p1 + 4);
  v16h r;
  r[0]  = (_Float16)x0.x; r[1]  = (_Float16)x0.y;
  r[2]  = (_Float16)x0.z; r[3]  = (_Float16)x0.w;
  r[4]  = (_Float16)x1.x; r[5]  = (_Float16)x1.y;
  r[6]  = (_Float16)x1.z; r[7]  = (_Float16)x1.w;
  r[8]  = (_Float16)x2.x; r[9]  = (_Float16)x2.y;
  r[10] = (_Float16)x2.z; r[11] = (_Float16)x2.w;
  r[12] = (_Float16)x3.x; r[13] = (_Float16)x3.y;
  r[14] = (_Float16)x3.z; r[15] = (_Float16)x3.w;
  return r;
}

// One wave accumulates a 16x16 f32 tile over K (multiple of 64 after unroll).
// A fragment (16x32 f16, ISA p.108): lane L = row M=L&15; dwords 0..3 hold
//   K = base..base+7, dwords 4..7 hold K = base+16..base+23, base = (L>=16)*8.
//   -> two contiguous 8-float runs: Arow+kb+hi*8 and +16.
// B fragment (32x16 f16): lane L = col N=L&15; same K layout as A:
//   two contiguous 8-float runs: Brow+kb+hi*8 and +16.
// Padded columns may alias any valid row: their products stay in lanes whose
// results are never stored (D col N depends only on B lanes N, N+16).
__device__ __forceinline__ v8f wmma_tile_acc(v8f acc,
                                             const float* __restrict__ Arow,
                                             const float* __restrict__ Brow,
                                             int K, int lane) {
  const int hi = lane >> 4;
#pragma unroll 2
  for (int kb = 0; kb < K; kb += 32) {
    const float* ap = Arow + kb + hi * 8;
    const float* bp = Brow + kb + hi * 8;
    const v16h a = cvt16(ap, ap + 16);
    const v16h b = cvt16(bp, bp + 16);
    // (neg_a, A, neg_b, B, c_mod, C, reuse_a, reuse_b)
    acc = __builtin_amdgcn_wmma_f32_16x16x32_f16(false, a, false, b,
                                                 (short)0, acc, false, false);
  }
  return acc;
}

__global__ void __launch_bounds__(128)
k_seq2sql(const float* __restrict__ wemb,
          const int* __restrict__ l_n, const int* __restrict__ l_hs,
          const int* __restrict__ start, const int* __restrict__ col,
          const int* __restrict__ ids,
          const float* __restrict__ Wss,  const float* __restrict__ bss,
          const float* __restrict__ Wse,  const float* __restrict__ bse,
          const float* __restrict__ Wsh,  const float* __restrict__ bsh,
          const float* __restrict__ Wwc,  const float* __restrict__ bwc,
          const float* __restrict__ Wsagg, const float* __restrict__ bsagg,
          const float* __restrict__ Wwop, const float* __restrict__ bwop,
          const float* __restrict__ Wconn, const float* __restrict__ bconn,
          const float* __restrict__ Wact, const float* __restrict__ bact,
          const float* __restrict__ Wslen, const float* __restrict__ bslen,
          const float* __restrict__ Wwlen, const float* __restrict__ bwlen,
          float* __restrict__ out) {
  const int lane = threadIdx.x;
  const int tile = blockIdx.x * 4 + threadIdx.y;   // wave-uniform job id
  const int m  = lane & 15;
  const int hi = lane >> 4;

  if (tile < T_QV) {
    // ---- qv: rows r=b*66+pos, gathered question tokens; cols ss|se --------
    const int r = tile * 16 + m;
    const int bb = r / 66, pos = r % 66;
    const int ln = l_n[ids[bb]];
    const int idx = (pos < ln + 2) ? pos : (ln + 1);
    const float* Arow = wemb + (size_t)(bb * LSEQ + idx) * ISZ;
    const int n = m;
    const float* Brow; float bias;
    if (n < 4)      { Brow = Wss + n * ISZ;       bias = bss[n]; }
    else if (n < 8) { Brow = Wse + (n - 4) * ISZ; bias = bse[n - 4]; }
    else            { Brow = Wss;                 bias = 0.0f; }   // pad, never stored
    v8f acc = {};
    acc = wmma_tile_acc(acc, Arow, Brow, ISZ, lane);
#pragma unroll
    for (int j = 0; j < 8; ++j) {
      const int r2 = tile * 16 + j + hi * 8;
      const int b2 = r2 / 66, p2 = r2 % 66;
      const float v = acc[j] + bias;
      if (n < 4)      out[OFF_WVSS + b2 * 264 + n * 66 + p2] = v;
      else if (n < 8) out[OFF_WVSE + b2 * 264 + (n - 4) * 66 + p2] = v;
    }
  } else if (tile < T_HT) {
    // ---- ht: rows r=b*32+pos, gathered header tokens; cols wcht|sht -------
    const int t = tile - T_QV;
    const int r = t * 16 + m;
    const int bb = r >> 5, pos = r & 31;
    const int id = ids[bb];
    int lhs = l_hs[id]; if (lhs < 1) lhs = 1;
    const int idx = col[id] + ((pos < lhs) ? pos : 0);
    const float* Arow = wemb + (size_t)(bb * LSEQ + idx) * ISZ;
    const int n = m;
    const float* Brow; float bias;
    if (n < 4)      { Brow = Wwc + n * ISZ;       bias = bwc[n]; }
    else if (n < 8) { Brow = Wsh + (n - 4) * ISZ; bias = bsh[n - 4]; }
    else            { Brow = Wwc;                 bias = 0.0f; }
    v8f acc = {};
    acc = wmma_tile_acc(acc, Arow, Brow, ISZ, lane);
#pragma unroll
    for (int j = 0; j < 8; ++j) {
      const int r2 = t * 16 + j + hi * 8;
      const int b2 = r2 >> 5, p2 = r2 & 31;
      const float v = acc[j] + bias;
      if (n < 4)      out[OFF_WC + b2 * 128 + n * 32 + p2] = v;
      else if (n < 8) out[OFF_SC + b2 * 128 + (n - 4) * 32 + p2] = v;
    }
  } else if (tile < T_SMALL) {
    // ---- small heads: rows (action,conn,wlen,slen)x64; 18 cols, 2 N-tiles -
    const int t = tile - T_HT;
    const int mt = t & 15, nt = t >> 4;
    const int r = mt * 16 + m;
    const int bb = r >> 2, rt = r & 3;
    const int s = start[ids[bb]];
    const int idx = (rt == 0) ? (s + 1) : (rt == 1) ? (s + 2)
                  : (rt == 2) ? (s + 3) : (s + 8);
    const float* Arow = wemb + (size_t)(bb * LSEQ + idx) * ISZ;
    const int n = nt * 16 + m;   // 0..4 act, 5..7 conn, 8..12 slen, 13..17 wlen
    const float* Brow; float bias;
    if (n < 5)       { Brow = Wact  + n * ISZ;        bias = bact[n]; }
    else if (n < 8)  { Brow = Wconn + (n - 5) * ISZ;  bias = bconn[n - 5]; }
    else if (n < 13) { Brow = Wslen + (n - 8) * ISZ;  bias = bslen[n - 8]; }
    else if (n < 18) { Brow = Wwlen + (n - 13) * ISZ; bias = bwlen[n - 13]; }
    else             { Brow = Wact;                   bias = 0.0f; }
    v8f acc = {};
    acc = wmma_tile_acc(acc, Arow, Brow, ISZ, lane);
#pragma unroll
    for (int j = 0; j < 8; ++j) {
      const int r2 = mt * 16 + j + hi * 8;
      const int b2 = r2 >> 2, rt2 = r2 & 3;
      const float v = acc[j] + bias;
      if (rt2 == 0 && n < 5)                  out[OFF_ACT  + b2 * 5 + n] = v;
      else if (rt2 == 1 && n >= 5  && n < 8)  out[OFF_CCO  + b2 * 3 + (n - 5)]  = v;
      else if (rt2 == 2 && n >= 13 && n < 18) out[OFF_WLEN + b2 * 5 + (n - 13)] = v;
      else if (rt2 == 3 && n >= 8  && n < 13) out[OFF_SLEN + b2 * 5 + (n - 8)]  = v;
    }
  } else if (tile < T_WO) {
    // ---- wo: row=batch, K=4x1024 concat of wo_emb rows; W_wop[16,4096] ----
    const int t = tile - T_SMALL;
    const int bb = t * 16 + m;
    const int s = start[ids[bb]];
    const int n = m;                                  // N = 16 exactly
    v8f acc = {};
#pragma unroll
    for (int i = 0; i < 4; ++i) {
      const float* Arow = wemb + (size_t)(bb * LSEQ + s + 4 + i) * ISZ;
      const float* Brow = Wwop + n * 4096 + i * ISZ;
      acc = wmma_tile_acc(acc, Arow, Brow, ISZ, lane);
    }
#pragma unroll
    for (int j = 0; j < 8; ++j) {
      const int b2 = t * 16 + j + hi * 8;
      out[OFF_WO + b2 * 16 + n] = acc[j] + bwop[n];
    }
  } else {
    // ---- sa: row=batch, K=4x1024 concat of sa_emb rows; W_sagg[24,4096] ---
    const int t = tile - T_WO;
    const int mt = t & 3, nt = t >> 2;
    const int bb = mt * 16 + m;
    const int s = start[ids[bb]];
    const int n = nt * 16 + m;                        // 24 real cols, pad 32
    v8f acc = {};
#pragma unroll
    for (int i = 0; i < 4; ++i) {
      const float* Arow = wemb + (size_t)(bb * LSEQ + s + 9 + i) * ISZ;
      const float* Brow = (n < 24) ? (Wsagg + n * 4096 + i * ISZ) : Wsagg;
      acc = wmma_tile_acc(acc, Arow, Brow, ISZ, lane);
    }
#pragma unroll
    for (int j = 0; j < 8; ++j) {
      const int b2 = mt * 16 + j + hi * 8;
      if (n < 24) out[OFF_SA + b2 * 24 + n] = acc[j] + bsagg[n];
    }
  }
}

extern "C" void kernel_launch(void* const* d_in, const int* in_sizes, int n_in,
                              void* d_out, int out_size, void* d_ws, size_t ws_size,
                              hipStream_t stream) {
  (void)in_sizes; (void)n_in; (void)out_size; (void)d_ws; (void)ws_size;
  const float* wemb   = (const float*)d_in[0];
  const int*   l_n    = (const int*)d_in[1];
  const int*   l_hs   = (const int*)d_in[2];
  const int*   start  = (const int*)d_in[3];
  const int*   col    = (const int*)d_in[4];
  const int*   ids    = (const int*)d_in[5];
  // d_in[6], d_in[7] = max_l_n, max_l_hs (compile-time constants here)
  const float* W_wsss = (const float*)d_in[8];
  const float* b_wsss = (const float*)d_in[9];
  const float* W_wsse = (const float*)d_in[10];
  const float* b_wsse = (const float*)d_in[11];
  const float* W_sht  = (const float*)d_in[12];
  const float* b_sht  = (const float*)d_in[13];
  const float* W_wcht = (const float*)d_in[14];
  const float* b_wcht = (const float*)d_in[15];
  const float* W_sagg = (const float*)d_in[16];
  const float* b_sagg = (const float*)d_in[17];
  const float* W_wop  = (const float*)d_in[18];
  const float* b_wop  = (const float*)d_in[19];
  const float* W_conn = (const float*)d_in[20];
  const float* b_conn = (const float*)d_in[21];
  const float* W_act  = (const float*)d_in[22];
  const float* b_act  = (const float*)d_in[23];
  const float* W_slen = (const float*)d_in[24];
  const float* b_slen = (const float*)d_in[25];
  const float* W_wlen = (const float*)d_in[26];
  const float* b_wlen = (const float*)d_in[27];
  float* out = (float*)d_out;

  // 436 wave tiles total, 4 wave32 per block -> 109 blocks, single launch
  hipLaunchKernelGGL(k_seq2sql, dim3(109), dim3(32, 4), 0, stream,
                     wemb, l_n, l_hs, start, col, ids,
                     W_wsss, b_wsss, W_wsse, b_wsse,
                     W_sht, b_sht, W_wcht, b_wcht,
                     W_sagg, b_sagg, W_wop, b_wop,
                     W_conn, b_conn, W_act, b_act,
                     W_slen, b_slen, W_wlen, b_wlen,
                     out);
}
